// Attention_1803886264479
// MI455X (gfx1250) — matrix operand order repaired and checked by reading, not compiled, not run on hardware
//
#include <hip/hip_runtime.h>
#include <hip/hip_bf16.h>

typedef __attribute__((ext_vector_type(16))) __bf16 v16bf;
typedef __attribute__((ext_vector_type(8)))  __bf16 v8bf;
typedef __attribute__((ext_vector_type(4)))  __bf16 v4bf;
typedef __attribute__((ext_vector_type(8)))  float  v8f;

#define B_   32
#define S_   4096
#define E_   512            // ENC_DIM == DEC_DIM == ATT_DIM
#define ROWS 32             // s-rows per workgroup in phase 1
#define NCHUNK (S_ / ROWS)  // 128
#define APAD 520            // bf16 elems per LDS row (512 + 8 pad -> 1040B stride)

// workspace layout (float offsets)
#define OFF_DBIAS 0                              // B*E
#define OFF_WT    (OFF_DBIAS + B_ * E_)          // Wt bf16[512][512] = 131072 float slots
#define OFF_ATT   (OFF_WT + (E_ * E_ / 2))       // B*S scores
#define OFF_PM    (OFF_ATT + B_ * S_)            // B*NCHUNK chunk max
#define OFF_PS    (OFF_PM + B_ * NCHUNK)         // B*NCHUNK chunk sum-exp
#define OFF_PW    (OFF_PS + B_ * NCHUNK)         // B*NCHUNK*E partial weighted
#define OFF_MZ    (OFF_PW + B_ * NCHUNK * E_)    // 2*B global max/Z

// ---------------- Phase 0a: dbias[b,a] = dec_h[b,:]@W_dec[:,a] + b_dec[a] + b_enc[a]
__global__ __launch_bounds__(128) void k_decbias(
    const float* __restrict__ dec_h, const float* __restrict__ W_dec,
    const float* __restrict__ b_dec, const float* __restrict__ b_enc,
    float* __restrict__ dbias) {
  int a = blockIdx.x * 128 + threadIdx.x;
  int b = blockIdx.y;
  float sum = b_dec[a] + b_enc[a];
  const float* dh = dec_h + b * E_;
  for (int e = 0; e < E_; ++e) sum = fmaf(dh[e], W_dec[e * E_ + a], sum);
  dbias[b * E_ + a] = sum;
}

// ---------------- Phase 0b: Wt[a][e] = bf16(W_enc[e][a])  (transposed bf16 copy, 512 KB)
__global__ __launch_bounds__(256) void k_wt(const float* __restrict__ W_enc,
                                            __bf16* __restrict__ Wt) {
  int idx = blockIdx.x * 256 + threadIdx.x;  // 0..262143
  int a = idx & (E_ - 1);
  int e = idx >> 9;
  Wt[a * E_ + e] = (__bf16)W_enc[e * E_ + a];  // coalesced reads, scattered 2B writes (1MB one-off)
}

// ---------------- Phase 1: fused scores + partial softmax + partial weighted sum
__global__ __launch_bounds__(64) void k_attn(
    const float* __restrict__ input, const __bf16* __restrict__ Wt,
    const float* __restrict__ dbias, const float* __restrict__ w_att,
    const float* __restrict__ b_att, float* __restrict__ att_g,
    float* __restrict__ part_m, float* __restrict__ part_s,
    float* __restrict__ part_w) {
  __shared__ __bf16 Ash[ROWS * APAD];
  __shared__ float attL[ROWS];
  __shared__ float pL[ROWS];

  const int tid = threadIdx.x;
  const int b = blockIdx.y;
  const int chunk = blockIdx.x;
  const int s0 = chunk * ROWS;

  // Stage input[b, s0:s0+32, :] f32 -> bf16 LDS (row stride APAD)
  {
    const float4* in4 = (const float4*)(input + (size_t)(b * S_ + s0) * E_);
    #pragma unroll 4
    for (int i = 0; i < 64; ++i) {          // 64 iters * 64 threads * float4 = 32*512
      int idx4 = tid + 64 * i;
      int r = idx4 >> 7;                    // /128 float4 per row
      int e4 = idx4 & 127;
      float4 v = in4[idx4];
      v4bf h = {(__bf16)v.x, (__bf16)v.y, (__bf16)v.z, (__bf16)v.w};
      *(v4bf*)(&Ash[r * APAD + e4 * 4]) = h;
    }
  }
  __syncthreads();

  const int lane = tid & 31;
  const int wave = tid >> 5;
  const int half = lane >> 4;   // half-wave select per WMMA layout
  const int lm = lane & 15;
  const int r0 = wave * 16;     // 16 s-rows per wave

  float acc[8] = {0.f, 0.f, 0.f, 0.f, 0.f, 0.f, 0.f, 0.f};
  const char* arow = (const char*)&Ash[(r0 + lm) * APAD];

  for (int n0 = 0; n0 < 32; ++n0) {         // 32 a-tiles of 16
    const int aIdx = n0 * 16 + lm;          // N = lane%16
    const char* brow = (const char*)(Wt + (size_t)aIdx * E_);
    v8f c = {0.f, 0.f, 0.f, 0.f, 0.f, 0.f, 0.f, 0.f};
    #pragma unroll
    for (int k0 = 0; k0 < 16; ++k0) {       // K = 512 in steps of 32
      // A 16x32 bf16 layout: V0-3 = K+{0..7|8..15}, V4-7 = K+{16..23|24..31}
      const char* ap = arow + k0 * 64 + half * 16;
      v8bf alo = *(const v8bf*)(ap);
      v8bf ahi = *(const v8bf*)(ap + 32);
      v16bf af = __builtin_shufflevector(alo, ahi, 0, 1, 2, 3, 4, 5, 6, 7, 8, 9,
                                         10, 11, 12, 13, 14, 15);
      // B 32x16 bf16 layout: V0-3 = K+{0..7|8..15}, V4-7 = K+{16..23|24..31}
      const char* bp = brow + k0 * 64 + half * 16;
      v8bf blo = *(const v8bf*)(bp);
      v8bf bhi = *(const v8bf*)(bp + 32);
      v16bf bf = __builtin_shufflevector(blo, bhi, 0, 1, 2, 3, 4, 5, 6, 7, 8, 9,
                                         10, 11, 12, 13, 14, 15);
      c = __builtin_amdgcn_wmma_f32_16x16x32_bf16(false, af, false, bf,
                                                  (short)0, c, false, false);
    }
    // epilogue: relu(enc + dbias) * w_att, accumulated over a
    float wa = w_att[aIdx];
    float db = dbias[b * E_ + aIdx];
    #pragma unroll
    for (int j = 0; j < 8; ++j) acc[j] += fmaxf(c[j] + db, 0.0f) * wa;
  }

  // reduce across the 16 N-lanes of each half-wave
  #pragma unroll
  for (int j = 0; j < 8; ++j) {
    float v = acc[j];
    v += __shfl_xor(v, 1, 32);
    v += __shfl_xor(v, 2, 32);
    v += __shfl_xor(v, 4, 32);
    v += __shfl_xor(v, 8, 32);
    acc[j] = v;
  }
  float batt = b_att[0];
  if (lm == 0) {
    #pragma unroll
    for (int j = 0; j < 8; ++j) {
      int row = r0 + half * 8 + j;          // C layout: M = j + 8*half
      float a = acc[j] + batt;
      attL[row] = a;
      att_g[(size_t)b * S_ + s0 + row] = a;
    }
  }
  __syncthreads();

  // chunk-local softmax partials
  float mx = -3.0e38f;
  #pragma unroll 8
  for (int i = 0; i < ROWS; ++i) mx = fmaxf(mx, attL[i]);
  if (tid < ROWS) pL[tid] = __expf(attL[tid] - mx);
  __syncthreads();
  float Z = 0.f;
  #pragma unroll 8
  for (int i = 0; i < ROWS; ++i) Z += pL[i];
  if (tid == 0) {
    part_m[b * NCHUNK + chunk] = mx;
    part_s[b * NCHUNK + chunk] = Z;
  }

  // partial weighted sum: reuse LDS bf16 tile (no second HBM pass over input)
  const int e0 = tid * 8;
  float w8[8] = {0.f, 0.f, 0.f, 0.f, 0.f, 0.f, 0.f, 0.f};
  for (int m = 0; m < ROWS; ++m) {
    float pm = pL[m];
    v8bf av = *(const v8bf*)(&Ash[m * APAD + e0]);
    #pragma unroll
    for (int j = 0; j < 8; ++j) w8[j] = fmaf(pm, (float)av[j], w8[j]);
  }
  float* pw = part_w + (size_t)(b * NCHUNK + chunk) * E_ + e0;
  #pragma unroll
  for (int j = 0; j < 8; ++j) pw[j] = w8[j];
}

// ---------------- Phase 2: merge chunk partials -> weighted[b, e], global (M, Z)
__global__ __launch_bounds__(256) void k_combine(
    const float* __restrict__ part_m, const float* __restrict__ part_s,
    const float* __restrict__ part_w, float* __restrict__ weighted,
    float* __restrict__ MZ) {
  int b = blockIdx.x;
  int tid = threadIdx.x;  // 256 threads, 2 columns each
  float M = -3.0e38f;
  for (int c = 0; c < NCHUNK; ++c) M = fmaxf(M, part_m[b * NCHUNK + c]);
  float Z = 0.f;
  for (int c = 0; c < NCHUNK; ++c)
    Z += part_s[b * NCHUNK + c] * __expf(part_m[b * NCHUNK + c] - M);
  int e = tid * 2;
  float a0 = 0.f, a1 = 0.f;
  for (int c = 0; c < NCHUNK; ++c) {
    float sc = __expf(part_m[b * NCHUNK + c] - M);
    const float* pw = part_w + (size_t)(b * NCHUNK + c) * E_ + e;
    a0 = fmaf(pw[0], sc, a0);
    a1 = fmaf(pw[1], sc, a1);
  }
  weighted[b * E_ + e] = a0 / Z;
  weighted[b * E_ + e + 1] = a1 / Z;
  if (tid == 0) {
    MZ[b * 2] = M;
    MZ[b * 2 + 1] = Z;
  }
}

// ---------------- Phase 3: prob[b,s] = exp(att - M_b) / Z_b
__global__ __launch_bounds__(256) void k_prob(const float* __restrict__ att_g,
                                              const float* __restrict__ MZ,
                                              float* __restrict__ prob) {
  int i = blockIdx.x * 256 + threadIdx.x;  // B*S
  int b = i >> 12;                          // / 4096
  prob[i] = __expf(att_g[i] - MZ[b * 2]) * (1.0f / MZ[b * 2 + 1]);
}

extern "C" void kernel_launch(void* const* d_in, const int* in_sizes, int n_in,
                              void* d_out, int out_size, void* d_ws, size_t ws_size,
                              hipStream_t stream) {
  const float* input = (const float*)d_in[0];
  const float* dec_h = (const float*)d_in[1];
  const float* W_enc = (const float*)d_in[2];
  const float* b_enc = (const float*)d_in[3];
  const float* W_dec = (const float*)d_in[4];
  const float* b_dec = (const float*)d_in[5];
  const float* w_att = (const float*)d_in[6];
  const float* b_att = (const float*)d_in[7];

  float* ws = (float*)d_ws;
  float* out_weighted = (float*)d_out;            // [B, 512]
  float* out_prob = (float*)d_out + B_ * E_;      // [B, S]

  k_decbias<<<dim3(E_ / 128, B_), 128, 0, stream>>>(dec_h, W_dec, b_dec, b_enc,
                                                    ws + OFF_DBIAS);
  k_wt<<<(E_ * E_) / 256, 256, 0, stream>>>(W_enc, (__bf16*)(ws + OFF_WT));
  k_attn<<<dim3(NCHUNK, B_), 64, 0, stream>>>(
      input, (const __bf16*)(ws + OFF_WT), ws + OFF_DBIAS, w_att, b_att,
      ws + OFF_ATT, ws + OFF_PM, ws + OFF_PS, ws + OFF_PW);
  k_combine<<<B_, 256, 0, stream>>>(ws + OFF_PM, ws + OFF_PS, ws + OFF_PW,
                                    out_weighted, ws + OFF_MZ);
  k_prob<<<(B_ * S_) / 256, 256, 0, stream>>>(ws + OFF_ATT, ws + OFF_MZ, out_prob);
}
